// Net_80384607911992
// MI455X (gfx1250) — matrix operand order repaired and checked by reading, not compiled, not run on hardware
//
#include <hip/hip_runtime.h>
#include <hip/hip_bf16.h>
#include <math.h>

// ---------------- problem constants ----------------
#define NN      20000      // nodes (multiple of 16)
#define NE      320000     // edges
#define NEL     (NE + NN)  // edges + self loops
#define NG      64         // graphs
#define FIN     373
#define SEQL    21
#define EPSV    1e-5f
#define CCOLS   696        // concat width: 128 + 384 + 64 + 120

typedef __attribute__((ext_vector_type(16))) __bf16 v16bf;
typedef __attribute__((ext_vector_type(8)))  __bf16 v8bf;
typedef __attribute__((ext_vector_type(8)))  float  v8f;

// ---------------- helpers ----------------
__device__ inline __bf16 f2bf(float f) {
  unsigned u = __float_as_uint(f);
  unsigned r = (u + 0x7FFFu + ((u >> 16) & 1u)) >> 16;
  unsigned short h = (unsigned short)r;
  __bf16 out;
  __builtin_memcpy(&out, &h, 2);
  return out;
}
__device__ inline unsigned fenc(float f) {
  unsigned u = __float_as_uint(f);
  return (u & 0x80000000u) ? ~u : (u | 0x80000000u);
}
__device__ inline float fdec(unsigned u) {
  u = (u & 0x80000000u) ? (u & 0x7FFFFFFFu) : ~u;
  return __uint_as_float(u);
}
__device__ inline void edge_sl(const int* __restrict__ src, const int* __restrict__ dst,
                               int idx, int& s, int& d) {
  if (idx < NE) { s = src[idx]; d = dst[idx]; } else { s = idx - NE; d = idx - NE; }
}

// ---------------- operand staging: f32 -> padded bf16 ----------------
// A [M,K] f32 row-major  ->  Abf [M,Kpad] bf16, zero-padded K tail.
__global__ void a2bf_kernel(const float* __restrict__ A, __bf16* __restrict__ out,
                            int M, int K, int Kpad) {
  long long total = (long long)M * Kpad;
  for (long long i = (long long)blockIdx.x * blockDim.x + threadIdx.x; i < total;
       i += (long long)gridDim.x * blockDim.x) {
    int m = (int)(i / Kpad), k = (int)(i % Kpad);
    out[i] = (k < K) ? f2bf(A[(size_t)m * K + k]) : f2bf(0.0f);
  }
}
// W [K,N] f32 row-major -> WMMA B-fragment order, padded to [Kpad, Npad]:
//   offset = ((kt*tilesN + nt)*32 + lane)*16 + e
//   k = kt*32 + (lane>=16 ? 16 : 0) + e ; n = nt*16 + (lane&15)
__global__ void w2bf_kernel(const float* __restrict__ W, __bf16* __restrict__ out,
                            int K, int N, int Kpad, int Npad) {
  int tilesN = Npad >> 4;
  long long total = (long long)Kpad * Npad;
  for (long long o = (long long)blockIdx.x * blockDim.x + threadIdx.x; o < total;
       o += (long long)gridDim.x * blockDim.x) {
    int e    = (int)(o & 15);
    int lane = (int)((o >> 4) & 31);
    long long rest = o >> 9;
    int nt = (int)(rest % tilesN);
    int kt = (int)(rest / tilesN);
    int k = kt * 32 + ((lane >> 4) << 4) + e;
    int n = nt * 16 + (lane & 15);
    out[o] = (k < K && n < N) ? f2bf(W[(size_t)k * N + n]) : f2bf(0.0f);
  }
}

// ---------------- WMMA GEMM: C[M,Npad] = act(Abf[M,Kpad] @ Wswz + bias) ----------------
// Requirements: M%16==0, Npad%64==0, Kpad%32==0 (guaranteed by staging).
// One wave computes a 16x64 output strip: 4 accumulators, A fragment reused 4x.
// Inner loop: 2x b128 A loads + 4x b256 B loads + 4x v_wmma per 32-K step. No predication.
__global__ __launch_bounds__(256)
void wmma_gemm_bf16_kernel(const __bf16* __restrict__ A, const __bf16* __restrict__ Ws,
                           const float* __restrict__ bias, float* __restrict__ C,
                           int M, int Npad, int Kpad, int act) {
  int wave = threadIdx.x >> 5;
  int lane = threadIdx.x & 31;
  int tilesN  = Npad >> 4;
  int tilesN4 = tilesN >> 2;
  int tileId = blockIdx.x * 8 + wave;
  if (tileId >= (M >> 4) * tilesN4) return;      // wave-uniform exit
  int tm  = tileId / tilesN4;
  int ntB = (tileId % tilesN4) << 2;

  int lo  = lane & 15;
  int hiA = (lane >> 4) << 4;                    // +16 K-elements for upper half-wave
  int row = (tm << 4) + lo;

  const __bf16* Ap = A + (size_t)row * Kpad + hiA;
  const __bf16* Bp = Ws + (((size_t)ntB) << 9) + ((size_t)lane << 4);
  size_t bStep = (size_t)tilesN << 9;            // advance one K-tile (32 rows)

  v8f acc0 = {}, acc1 = {}, acc2 = {}, acc3 = {};
  int nk = Kpad >> 5;
  for (int kt = 0; kt < nk; ++kt) {
    v8bf a0 = *(const v8bf*)(Ap);
    v8bf a1 = *(const v8bf*)(Ap + 8);
    v16bf a = __builtin_shufflevector(a0, a1, 0, 1, 2, 3, 4, 5, 6, 7,
                                      8, 9, 10, 11, 12, 13, 14, 15);
    v16bf b0 = *(const v16bf*)(Bp);
    v16bf b1 = *(const v16bf*)(Bp + 512);
    v16bf b2 = *(const v16bf*)(Bp + 1024);
    v16bf b3 = *(const v16bf*)(Bp + 1536);
    acc0 = __builtin_amdgcn_wmma_f32_16x16x32_bf16(false, a, false, b0, (short)0, acc0, false, false);
    acc1 = __builtin_amdgcn_wmma_f32_16x16x32_bf16(false, a, false, b1, (short)0, acc1, false, false);
    acc2 = __builtin_amdgcn_wmma_f32_16x16x32_bf16(false, a, false, b2, (short)0, acc2, false, false);
    acc3 = __builtin_amdgcn_wmma_f32_16x16x32_bf16(false, a, false, b3, (short)0, acc3, false, false);
    Ap += 32;
    Bp += bStep;
  }

  int mrow = (tm << 4) + ((lane >> 4) << 3);     // C/D: upper lanes hold M+8
  int colB = (ntB << 4) + lo;
#pragma unroll
  for (int t = 0; t < 4; ++t) {
    v8f acc = (t == 0) ? acc0 : (t == 1) ? acc1 : (t == 2) ? acc2 : acc3;
    int col = colB + (t << 4);
    float bi = bias ? bias[col] : 0.0f;
#pragma unroll
    for (int r = 0; r < 8; ++r) {
      float v = acc[r] + bi;
      if (act) v = fmaxf(v, 0.0f);
      C[(size_t)(mrow + r) * Npad + col] = v;
    }
  }
}

// ---------------- elementwise / fill ----------------
__global__ void fill_f32_kernel(float* __restrict__ p, float v, int n) {
  int i = blockIdx.x * blockDim.x + threadIdx.x;
  if (i < n) p[i] = v;
}
__global__ void fill_u32_kernel(unsigned* __restrict__ p, unsigned v, int n) {
  int i = blockIdx.x * blockDim.x + threadIdx.x;
  if (i < n) p[i] = v;
}
__global__ void copy_f32_kernel(const float* __restrict__ a, float* __restrict__ b, int n) {
  for (int i = blockIdx.x * blockDim.x + threadIdx.x; i < n; i += gridDim.x * blockDim.x)
    b[i] = a[i];
}
__global__ void rsqrt_kernel(float* __restrict__ p, int n) {
  int i = blockIdx.x * blockDim.x + threadIdx.x;
  if (i < n) p[i] = rsqrtf(p[i]);
}
__global__ void deg_kernel(const int* __restrict__ dst, float* __restrict__ deg) {
  for (int e = blockIdx.x * blockDim.x + threadIdx.x; e < NE; e += gridDim.x * blockDim.x)
    atomicAdd(&deg[dst[e]], 1.0f);
}

// ---------------- GCN aggregation ----------------
__global__ void gcn_init_kernel(const float* __restrict__ t, const float* __restrict__ dinv,
                                const float* __restrict__ bias, float* __restrict__ out, int C) {
  long long total = (long long)NN * C;
  for (long long i = (long long)blockIdx.x * blockDim.x + threadIdx.x; i < total;
       i += (long long)gridDim.x * blockDim.x) {
    int n = (int)(i / C), c = (int)(i % C);
    float dn = dinv[n];
    out[i] = bias[c] + t[i] * dn * dn;       // self-loop term + bias
  }
}
__global__ void gcn_scatter_kernel(const float* __restrict__ t, const int* __restrict__ src,
                                   const int* __restrict__ dst, const float* __restrict__ dinv,
                                   float* __restrict__ out, int C) {
  long long total = (long long)NE * C;
  for (long long i = (long long)blockIdx.x * blockDim.x + threadIdx.x; i < total;
       i += (long long)gridDim.x * blockDim.x) {
    int e = (int)(i / C), c = (int)(i % C);
    int s = src[e], d = dst[e];
    atomicAdd(&out[(size_t)d * C + c], t[(size_t)s * C + c] * dinv[s] * dinv[d]);
  }
}

// ---------------- GIN aggregation ----------------
__global__ void gin_scatter_kernel(const float* __restrict__ h, const int* __restrict__ src,
                                   const int* __restrict__ dst, float* __restrict__ agg, int C) {
  long long total = (long long)NE * C;
  for (long long i = (long long)blockIdx.x * blockDim.x + threadIdx.x; i < total;
       i += (long long)gridDim.x * blockDim.x) {
    int e = (int)(i / C), c = (int)(i % C);
    atomicAdd(&agg[(size_t)dst[e] * C + c], h[(size_t)src[e] * C + c]);
  }
}

// ---------------- GAT attention ----------------
__global__ void att_logits_kernel(const float* __restrict__ hw, const float* __restrict__ asrc,
                                  const float* __restrict__ adst, float* __restrict__ alS,
                                  float* __restrict__ alD, int ch) {
  int t = blockIdx.x * blockDim.x + threadIdx.x;
  if (t >= NN * 3) return;
  int head = t % 3, n = t / 3;
  const float* hp = hw + (size_t)n * 3 * ch + (size_t)head * ch;
  float ss = 0.f, sd = 0.f;
  for (int c = 0; c < ch; ++c) {
    float v = hp[c];
    ss += v * asrc[head * ch + c];
    sd += v * adst[head * ch + c];
  }
  alS[t] = ss; alD[t] = sd;
}
__global__ void att_max_kernel(const int* __restrict__ src, const int* __restrict__ dst,
                               const float* __restrict__ alS, const float* __restrict__ alD,
                               unsigned* __restrict__ emax) {
  int total = NEL * 3;
  for (int t = blockIdx.x * blockDim.x + threadIdx.x; t < total; t += gridDim.x * blockDim.x) {
    int head = t % 3, idx = t / 3, s, d;
    edge_sl(src, dst, idx, s, d);
    float e = alS[s * 3 + head] + alD[d * 3 + head];
    e = e > 0.f ? e : 0.2f * e;  // leaky_relu
    atomicMax(&emax[d * 3 + head], fenc(e));
  }
}
__global__ void att_den_kernel(const int* __restrict__ src, const int* __restrict__ dst,
                               const float* __restrict__ alS, const float* __restrict__ alD,
                               const unsigned* __restrict__ emax, float* __restrict__ den) {
  int total = NEL * 3;
  for (int t = blockIdx.x * blockDim.x + threadIdx.x; t < total; t += gridDim.x * blockDim.x) {
    int head = t % 3, idx = t / 3, s, d;
    edge_sl(src, dst, idx, s, d);
    float e = alS[s * 3 + head] + alD[d * 3 + head];
    e = e > 0.f ? e : 0.2f * e;
    atomicAdd(&den[d * 3 + head], __expf(e - fdec(emax[d * 3 + head])));
  }
}
__global__ void out_init_bias_kernel(float* __restrict__ out, const float* __restrict__ bias, int C) {
  long long total = (long long)NN * C;
  for (long long i = (long long)blockIdx.x * blockDim.x + threadIdx.x; i < total;
       i += (long long)gridDim.x * blockDim.x)
    out[i] = bias[(int)(i % C)];
}
__global__ void att_scatter_kernel(const float* __restrict__ hw, const int* __restrict__ src,
                                   const int* __restrict__ dst, const float* __restrict__ alS,
                                   const float* __restrict__ alD, const unsigned* __restrict__ emax,
                                   const float* __restrict__ den, float* __restrict__ out, int ch) {
  int total = NEL * 3;
  int C = 3 * ch;
  for (int t = blockIdx.x * blockDim.x + threadIdx.x; t < total; t += gridDim.x * blockDim.x) {
    int head = t % 3, idx = t / 3, s, d;
    edge_sl(src, dst, idx, s, d);
    float e = alS[s * 3 + head] + alD[d * 3 + head];
    e = e > 0.f ? e : 0.2f * e;
    float alpha = __expf(e - fdec(emax[d * 3 + head])) / den[d * 3 + head];
    const float* hs = hw + (size_t)s * C + (size_t)head * ch;
    float* od = out + (size_t)d * C + (size_t)head * ch;
    for (int c = 0; c < ch; ++c) atomicAdd(&od[c], hs[c] * alpha);
  }
}

// ---------------- training-mode BatchNorm over rows, fused preceding ReLU ----------------
__global__ void bn_relu_kernel(float* __restrict__ h, const float* __restrict__ g,
                               const float* __restrict__ b, int Nn, int C) {
  int c = blockIdx.x;              // one block per channel
  int tid = threadIdx.x;
  __shared__ float s1[256], s2[256];
  float sum = 0.f, sq = 0.f;
  for (int i = tid; i < Nn; i += 256) {
    float v = fmaxf(h[(size_t)i * C + c], 0.f);
    sum += v; sq += v * v;
  }
  s1[tid] = sum; s2[tid] = sq;
  __syncthreads();
  for (int s = 128; s > 0; s >>= 1) {
    if (tid < s) { s1[tid] += s1[tid + s]; s2[tid] += s2[tid + s]; }
    __syncthreads();
  }
  float mean = s1[0] / (float)Nn;
  float var  = s2[0] / (float)Nn - mean * mean;
  float scale = g[c] * rsqrtf(var + EPSV);
  float shift = b[c] - mean * scale;
  for (int i = tid; i < Nn; i += 256) {
    size_t o = (size_t)i * C + c;
    h[o] = fmaxf(h[o], 0.f) * scale + shift;
  }
}

// BatchNorm over [B, C, L] (mean/var over B and L), fused ReLU, in place
__global__ void bn_nc_relu_kernel(float* __restrict__ h, const float* __restrict__ g,
                                  const float* __restrict__ b, int B, int C, int L) {
  int c = blockIdx.x;
  int tid = threadIdx.x;
  __shared__ float s1[256], s2[256];
  int tot = B * L;
  float sum = 0.f, sq = 0.f;
  for (int i = tid; i < tot; i += 256) {
    int bb = i / L, l = i % L;
    float v = fmaxf(h[(size_t)bb * C * L + (size_t)c * L + l], 0.f);
    sum += v; sq += v * v;
  }
  s1[tid] = sum; s2[tid] = sq;
  __syncthreads();
  for (int s = 128; s > 0; s >>= 1) {
    if (tid < s) { s1[tid] += s1[tid + s]; s2[tid] += s2[tid + s]; }
    __syncthreads();
  }
  float mean = s1[0] / (float)tot;
  float var  = s2[0] / (float)tot - mean * mean;
  float scale = g[c] * rsqrtf(var + EPSV);
  float shift = b[c] - mean * scale;
  for (int i = tid; i < tot; i += 256) {
    int bb = i / L, l = i % L;
    size_t o = (size_t)bb * C * L + (size_t)c * L + l;
    h[o] = fmaxf(h[o], 0.f) * scale + shift;
  }
}

// ---------------- pooling (segment_sum over batch) into concat columns ----------------
__global__ void pool_kernel(const float* __restrict__ h, const int* __restrict__ batch,
                            float* __restrict__ concat, int C, int colOff) {
  long long total = (long long)NN * C;
  for (long long i = (long long)blockIdx.x * blockDim.x + threadIdx.x; i < total;
       i += (long long)gridDim.x * blockDim.x) {
    int n = (int)(i / C), c = (int)(i % C);
    atomicAdd(&concat[(size_t)batch[n] * CCOLS + colOff + c], h[i]);
  }
}

// ---------------- 1D conv (VALID, k=3) ----------------
__global__ void conv1d_kernel(const float* __restrict__ in, const float* __restrict__ W,
                              const float* __restrict__ bias, float* __restrict__ out,
                              int B, int Cin, int Lin, int Cout) {
  int Lout = Lin - 2;
  int t = blockIdx.x * blockDim.x + threadIdx.x;
  int total = B * Cout * Lout;
  if (t >= total) return;
  int l = t % Lout, co = (t / Lout) % Cout, bb = t / (Lout * Cout);
  float acc = bias[co];
  for (int ci = 0; ci < Cin; ++ci)
#pragma unroll
    for (int k = 0; k < 3; ++k)
      acc += in[(size_t)bb * Cin * Lin + (size_t)ci * Lin + l + k] * W[(size_t)co * Cin * 3 + ci * 3 + k];
  out[(size_t)bb * Cout * Lout + (size_t)co * Lout + l] = acc;
}
__global__ void sc_copy_kernel(const float* __restrict__ sc, float* __restrict__ concat) {
  int t = blockIdx.x * blockDim.x + threadIdx.x;
  if (t >= NG * 120) return;
  int bb = t / 120, i = t % 120;
  concat[(size_t)bb * CCOLS + 576 + i] = sc[t];
}
__global__ void fc3_out_kernel(const float* __restrict__ f3p, const float* __restrict__ b,
                               float* __restrict__ out) {
  int g = blockIdx.x * blockDim.x + threadIdx.x;
  if (g < NG) out[g] = fmaxf(f3p[(size_t)g * 64] + b[0], 0.0f);
}

// ---------------- host-side helpers ----------------
static inline int kpad32(int k) { return (k + 31) & ~31; }

static void launch_wmma(const __bf16* Abf, const __bf16* Ws, const float* bias, float* C,
                        int M, int Npad, int Kpad, int act, hipStream_t st) {
  int tiles = (M / 16) * (Npad / 64);
  int blocks = (tiles + 7) / 8;
  wmma_gemm_bf16_kernel<<<blocks, 256, 0, st>>>(Abf, Ws, bias, C, M, Npad, Kpad, act);
}
// convert f32 A then GEMM
static void gemm(const float* A, const __bf16* Ws, const float* bias, float* C,
                 int M, int Npad, int K, int act, __bf16* Abf, hipStream_t st) {
  int Kpad = kpad32(K);
  long long tot = (long long)M * Kpad;
  int blocks = (int)((tot + 255) / 256); if (blocks > 8192) blocks = 8192;
  a2bf_kernel<<<blocks, 256, 0, st>>>(A, Abf, M, K, Kpad);
  launch_wmma(Abf, Ws, bias, C, M, Npad, Kpad, act, st);
}

extern "C" void kernel_launch(void* const* d_in, const int* in_sizes, int n_in,
                              void* d_out, int out_size, void* d_ws, size_t ws_size,
                              hipStream_t stream) {
  (void)in_sizes; (void)n_in; (void)out_size; (void)ws_size;
  const float* x     = (const float*)d_in[0];
  const int*   ei    = (const int*)d_in[1];
  const int*   src   = ei;
  const int*   dst   = ei + NE;
  const int*   batch = (const int*)d_in[2];
  const float* sIn   = (const float*)d_in[3];
  // params in insertion order of setup_inputs()["params"]
  int i = 4;
  const float* gcn1_W = (const float*)d_in[i++]; const float* gcn1_b = (const float*)d_in[i++];
  const float* gcn2_W = (const float*)d_in[i++]; const float* gcn2_b = (const float*)d_in[i++];
  const float* gcn3_W = (const float*)d_in[i++]; const float* gcn3_b = (const float*)d_in[i++];
  const float* gat1_W = (const float*)d_in[i++]; const float* gat1_as = (const float*)d_in[i++];
  const float* gat1_ad = (const float*)d_in[i++]; const float* gat1_b = (const float*)d_in[i++];
  const float* gat2_W = (const float*)d_in[i++]; const float* gat2_as = (const float*)d_in[i++];
  const float* gat2_ad = (const float*)d_in[i++]; const float* gat2_b = (const float*)d_in[i++];
  const float* gat3_W = (const float*)d_in[i++]; const float* gat3_as = (const float*)d_in[i++];
  const float* gat3_ad = (const float*)d_in[i++]; const float* gat3_b = (const float*)d_in[i++];
  const float* gin1_W1 = (const float*)d_in[i++]; const float* gin1_b1 = (const float*)d_in[i++];
  const float* gin1_W2 = (const float*)d_in[i++]; const float* gin1_b2 = (const float*)d_in[i++];
  const float* gin2_W1 = (const float*)d_in[i++]; const float* gin2_b1 = (const float*)d_in[i++];
  const float* gin2_W2 = (const float*)d_in[i++]; const float* gin2_b2 = (const float*)d_in[i++];
  const float* gin3_W1 = (const float*)d_in[i++]; const float* gin3_b1 = (const float*)d_in[i++];
  const float* gin3_W2 = (const float*)d_in[i++]; const float* gin3_b2 = (const float*)d_in[i++];
  const float* c1_W = (const float*)d_in[i++]; const float* c1_b = (const float*)d_in[i++];
  const float* c2_W = (const float*)d_in[i++]; const float* c2_b = (const float*)d_in[i++];
  const float* c3_W = (const float*)d_in[i++]; const float* c3_b = (const float*)d_in[i++];
  const float* bn01_g = (const float*)d_in[i++]; const float* bn01_b = (const float*)d_in[i++];
  const float* bn02_g = (const float*)d_in[i++]; const float* bn02_b = (const float*)d_in[i++];
  const float* bn03_g = (const float*)d_in[i++]; const float* bn03_b = (const float*)d_in[i++];
  const float* bn11_g = (const float*)d_in[i++]; const float* bn11_b = (const float*)d_in[i++];
  const float* bn12_g = (const float*)d_in[i++]; const float* bn12_b = (const float*)d_in[i++];
  const float* bn13_g = (const float*)d_in[i++]; const float* bn13_b = (const float*)d_in[i++];
  const float* bn21_g = (const float*)d_in[i++]; const float* bn21_b = (const float*)d_in[i++];
  const float* bn22_g = (const float*)d_in[i++]; const float* bn22_b = (const float*)d_in[i++];
  const float* bn23_g = (const float*)d_in[i++]; const float* bn23_b = (const float*)d_in[i++];
  const float* ba1_g = (const float*)d_in[i++]; const float* ba1_b = (const float*)d_in[i++];
  const float* ba2_g = (const float*)d_in[i++]; const float* ba2_b = (const float*)d_in[i++];
  const float* ba3_g = (const float*)d_in[i++]; const float* ba3_b = (const float*)d_in[i++];
  const float* fc1_W = (const float*)d_in[i++]; const float* fc1_b = (const float*)d_in[i++];
  const float* fc2_W = (const float*)d_in[i++]; const float* fc2_b = (const float*)d_in[i++];
  const float* fc3_W = (const float*)d_in[i++]; const float* fc3_b = (const float*)d_in[i++];

  // ---------------- workspace carve-up (float units, 64B-aligned chunks) ----------------
  const size_t BIG = (size_t)NN * 768;
  float* ws = (float*)d_ws;
  float* B0 = ws;
  float* B1 = ws + BIG;
  float* B2 = ws + 2 * BIG;
  float* sm = ws + 3 * BIG;
  float* dinv = sm;           sm += NN;
  float* alS  = sm;           sm += 3 * NN;
  float* alD  = sm;           sm += 3 * NN;
  unsigned* emax = (unsigned*)sm; sm += 3 * NN;
  float* den  = sm;           sm += 3 * NN;
  float* concat = sm;         sm += NG * CCOLS;
  float* cv1 = sm;            sm += NG * 16 * 19;
  float* cv2 = sm;            sm += NG * 32 * 17;
  float* cv3 = sm;            sm += NG * 8 * 15;
  float* f1  = sm;            sm += NG * 256;
  float* f2  = sm;            sm += NG * 64;
  float* f3p = sm;            sm += NG * 64;
  sm = (float*)(((uintptr_t)sm + 63) & ~(uintptr_t)63);
  __bf16* Abf = (__bf16*)sm;  sm += BIG;                 // bf16 staging for activations (2 bf16/float)
  __bf16* xbf = (__bf16*)sm;  sm += (size_t)NN * 384 / 2; // x pre-converted [NN, 384]
  __bf16* wArena = (__bf16*)sm;

  // weight arena offsets (bf16 elements, Kpad x Npad each)
  size_t wo = 0;
  __bf16* w_gcn1 = wArena + wo; wo += (size_t)384 * 256;
  __bf16* w_gcn2 = wArena + wo; wo += (size_t)256 * 128;
  __bf16* w_gcn3 = wArena + wo; wo += (size_t)128 * 128;
  __bf16* w_gat1 = wArena + wo; wo += (size_t)384 * 768;
  __bf16* w_gat2 = wArena + wo; wo += (size_t)768 * 384;
  __bf16* w_gat3 = wArena + wo; wo += (size_t)384 * 384;
  __bf16* w_gi1a = wArena + wo; wo += (size_t)384 * 256;
  __bf16* w_gi1b = wArena + wo; wo += (size_t)256 * 256;
  __bf16* w_gi2a = wArena + wo; wo += (size_t)256 * 128;
  __bf16* w_gi2b = wArena + wo; wo += (size_t)128 * 128;
  __bf16* w_gi3a = wArena + wo; wo += (size_t)128 * 64;
  __bf16* w_gi3b = wArena + wo; wo += (size_t)64 * 64;
  __bf16* w_fc1  = wArena + wo; wo += (size_t)704 * 256;
  __bf16* w_fc2  = wArena + wo; wo += (size_t)256 * 64;
  __bf16* w_fc3  = wArena + wo; wo += (size_t)64 * 64;

  const int T = 256;
  const int GS = 8192;   // grid-stride blocks for big scatters
  const int GA = 2048;   // for edge x head kernels
  hipStream_t st = stream;

  // ---------------- stage weights (swizzled bf16) and x ----------------
  w2bf_kernel<<<GS, T, 0, st>>>(gcn1_W, w_gcn1, FIN, 256, 384, 256);
  w2bf_kernel<<<GS, T, 0, st>>>(gcn2_W, w_gcn2, 256, 128, 256, 128);
  w2bf_kernel<<<GS, T, 0, st>>>(gcn3_W, w_gcn3, 128, 128, 128, 128);
  w2bf_kernel<<<GS, T, 0, st>>>(gat1_W, w_gat1, FIN, 768, 384, 768);
  w2bf_kernel<<<GS, T, 0, st>>>(gat2_W, w_gat2, 768, 384, 768, 384);
  w2bf_kernel<<<GS, T, 0, st>>>(gat3_W, w_gat3, 384, 384, 384, 384);
  w2bf_kernel<<<GS, T, 0, st>>>(gin1_W1, w_gi1a, FIN, 256, 384, 256);
  w2bf_kernel<<<GS, T, 0, st>>>(gin1_W2, w_gi1b, 256, 256, 256, 256);
  w2bf_kernel<<<GS, T, 0, st>>>(gin2_W1, w_gi2a, 256, 128, 256, 128);
  w2bf_kernel<<<GS, T, 0, st>>>(gin2_W2, w_gi2b, 128, 128, 128, 128);
  w2bf_kernel<<<GS, T, 0, st>>>(gin3_W1, w_gi3a, 128, 64, 128, 64);
  w2bf_kernel<<<GS, T, 0, st>>>(gin3_W2, w_gi3b, 64, 64, 64, 64);
  w2bf_kernel<<<GS, T, 0, st>>>(fc1_W, w_fc1, CCOLS, 256, 704, 256);
  w2bf_kernel<<<GS, T, 0, st>>>(fc2_W, w_fc2, 256, 64, 256, 64);
  w2bf_kernel<<<GS, T, 0, st>>>(fc3_W, w_fc3, 64, 1, 64, 64);
  a2bf_kernel<<<GS, T, 0, st>>>(x, xbf, NN, FIN, 384);

  // degrees (self-loop counted by init=1.0), then dinv = rsqrt(deg)
  fill_f32_kernel<<<(NN + T - 1) / T, T, 0, st>>>(dinv, 1.0f, NN);
  deg_kernel<<<GA, T, 0, st>>>(dst, dinv);
  rsqrt_kernel<<<(NN + T - 1) / T, T, 0, st>>>(dinv, NN);

  fill_f32_kernel<<<(NG * CCOLS + T - 1) / T, T, 0, st>>>(concat, 0.0f, NG * CCOLS);

  // ================= GCN branch =================
  launch_wmma(xbf, w_gcn1, nullptr, B0, NN, 256, 384, 0, st);
  gcn_init_kernel<<<GS, T, 0, st>>>(B0, dinv, gcn1_b, B1, 256);
  gcn_scatter_kernel<<<GS, T, 0, st>>>(B0, src, dst, dinv, B1, 256);
  bn_relu_kernel<<<256, 256, 0, st>>>(B1, bn01_g, bn01_b, NN, 256);
  gemm(B1, w_gcn2, nullptr, B0, NN, 128, 256, 0, Abf, st);
  gcn_init_kernel<<<GS, T, 0, st>>>(B0, dinv, gcn2_b, B1, 128);
  gcn_scatter_kernel<<<GS, T, 0, st>>>(B0, src, dst, dinv, B1, 128);
  bn_relu_kernel<<<128, 256, 0, st>>>(B1, bn02_g, bn02_b, NN, 128);
  gemm(B1, w_gcn3, nullptr, B0, NN, 128, 128, 0, Abf, st);
  gcn_init_kernel<<<GS, T, 0, st>>>(B0, dinv, gcn3_b, B1, 128);
  gcn_scatter_kernel<<<GS, T, 0, st>>>(B0, src, dst, dinv, B1, 128);
  bn_relu_kernel<<<128, 256, 0, st>>>(B1, bn03_g, bn03_b, NN, 128);
  pool_kernel<<<GS, T, 0, st>>>(B1, batch, concat, 128, 0);

  // ================= GAT branch =================
  // layer 1: FIN -> 3x256
  launch_wmma(xbf, w_gat1, nullptr, B0, NN, 768, 384, 0, st);
  att_logits_kernel<<<(NN * 3 + T - 1) / T, T, 0, st>>>(B0, gat1_as, gat1_ad, alS, alD, 256);
  fill_u32_kernel<<<(3 * NN + T - 1) / T, T, 0, st>>>(emax, 0u, 3 * NN);
  att_max_kernel<<<GA, T, 0, st>>>(src, dst, alS, alD, emax);
  fill_f32_kernel<<<(3 * NN + T - 1) / T, T, 0, st>>>(den, 0.0f, 3 * NN);
  att_den_kernel<<<GA, T, 0, st>>>(src, dst, alS, alD, emax, den);
  out_init_bias_kernel<<<GS, T, 0, st>>>(B1, gat1_b, 768);
  att_scatter_kernel<<<GA, T, 0, st>>>(B0, src, dst, alS, alD, emax, den, B1, 256);
  bn_relu_kernel<<<768, 256, 0, st>>>(B1, bn11_g, bn11_b, NN, 768);
  // layer 2: 768 -> 3x128
  gemm(B1, w_gat2, nullptr, B0, NN, 384, 768, 0, Abf, st);
  att_logits_kernel<<<(NN * 3 + T - 1) / T, T, 0, st>>>(B0, gat2_as, gat2_ad, alS, alD, 128);
  fill_u32_kernel<<<(3 * NN + T - 1) / T, T, 0, st>>>(emax, 0u, 3 * NN);
  att_max_kernel<<<GA, T, 0, st>>>(src, dst, alS, alD, emax);
  fill_f32_kernel<<<(3 * NN + T - 1) / T, T, 0, st>>>(den, 0.0f, 3 * NN);
  att_den_kernel<<<GA, T, 0, st>>>(src, dst, alS, alD, emax, den);
  out_init_bias_kernel<<<GS, T, 0, st>>>(B2, gat2_b, 384);
  att_scatter_kernel<<<GA, T, 0, st>>>(B0, src, dst, alS, alD, emax, den, B2, 128);
  bn_relu_kernel<<<384, 256, 0, st>>>(B2, bn12_g, bn12_b, NN, 384);
  // layer 3: 384 -> 3x128
  gemm(B2, w_gat3, nullptr, B0, NN, 384, 384, 0, Abf, st);
  att_logits_kernel<<<(NN * 3 + T - 1) / T, T, 0, st>>>(B0, gat3_as, gat3_ad, alS, alD, 128);
  fill_u32_kernel<<<(3 * NN + T - 1) / T, T, 0, st>>>(emax, 0u, 3 * NN);
  att_max_kernel<<<GA, T, 0, st>>>(src, dst, alS, alD, emax);
  fill_f32_kernel<<<(3 * NN + T - 1) / T, T, 0, st>>>(den, 0.0f, 3 * NN);
  att_den_kernel<<<GA, T, 0, st>>>(src, dst, alS, alD, emax, den);
  out_init_bias_kernel<<<GS, T, 0, st>>>(B1, gat3_b, 384);
  att_scatter_kernel<<<GA, T, 0, st>>>(B0, src, dst, alS, alD, emax, den, B1, 128);
  bn_relu_kernel<<<384, 256, 0, st>>>(B1, bn13_g, bn13_b, NN, 384);
  pool_kernel<<<GS, T, 0, st>>>(B1, batch, concat, 384, 128);

  // ================= GIN branch =================
  copy_f32_kernel<<<GS, T, 0, st>>>(x, B0, NN * FIN);
  gin_scatter_kernel<<<GS, T, 0, st>>>(x, src, dst, B0, FIN);
  gemm(B0, w_gi1a, gin1_b1, B1, NN, 256, FIN, 1, Abf, st);
  gemm(B1, w_gi1b, gin1_b2, B2, NN, 256, 256, 0, Abf, st);
  bn_relu_kernel<<<256, 256, 0, st>>>(B2, bn21_g, bn21_b, NN, 256);
  copy_f32_kernel<<<GS, T, 0, st>>>(B2, B0, NN * 256);
  gin_scatter_kernel<<<GS, T, 0, st>>>(B2, src, dst, B0, 256);
  gemm(B0, w_gi2a, gin2_b1, B1, NN, 128, 256, 1, Abf, st);
  gemm(B1, w_gi2b, gin2_b2, B2, NN, 128, 128, 0, Abf, st);
  bn_relu_kernel<<<128, 256, 0, st>>>(B2, bn22_g, bn22_b, NN, 128);
  copy_f32_kernel<<<GS, T, 0, st>>>(B2, B0, NN * 128);
  gin_scatter_kernel<<<GS, T, 0, st>>>(B2, src, dst, B0, 128);
  gemm(B0, w_gi3a, gin3_b1, B1, NN, 64, 128, 1, Abf, st);
  gemm(B1, w_gi3b, gin3_b2, B2, NN, 64, 64, 0, Abf, st);
  bn_relu_kernel<<<64, 256, 0, st>>>(B2, bn23_g, bn23_b, NN, 64);
  pool_kernel<<<GS, T, 0, st>>>(B2, batch, concat, 64, 512);

  // ================= 1D CNN branch =================
  conv1d_kernel<<<(NG * 16 * 19 + T - 1) / T, T, 0, st>>>(sIn, c1_W, c1_b, cv1, NG, 3, SEQL, 16);
  bn_nc_relu_kernel<<<16, 256, 0, st>>>(cv1, ba1_g, ba1_b, NG, 16, 19);
  conv1d_kernel<<<(NG * 32 * 17 + T - 1) / T, T, 0, st>>>(cv1, c2_W, c2_b, cv2, NG, 16, 19, 32);
  bn_nc_relu_kernel<<<32, 256, 0, st>>>(cv2, ba2_g, ba2_b, NG, 32, 17);
  conv1d_kernel<<<(NG * 8 * 15 + T - 1) / T, T, 0, st>>>(cv2, c3_W, c3_b, cv3, NG, 32, 17, 8);
  bn_nc_relu_kernel<<<8, 256, 0, st>>>(cv3, ba3_g, ba3_b, NG, 8, 15);
  sc_copy_kernel<<<(NG * 120 + T - 1) / T, T, 0, st>>>(cv3, concat);

  // ================= FC head =================
  gemm(concat, w_fc1, fc1_b, f1, NG, 256, CCOLS, 1, Abf, st);
  gemm(f1, w_fc2, fc2_b, f2, NG, 64, 256, 1, Abf, st);
  gemm(f2, w_fc3, nullptr, f3p, NG, 64, 64, 0, Abf, st);   // N padded 1->64
  fc3_out_kernel<<<1, 64, 0, st>>>(f3p, fc3_b, (float*)d_out);
}
